// BiLSTMCRFModel_17403207483845
// MI455X (gfx1250) — matrix operand order repaired and checked by reading, not compiled, not run on hardware
//
#include <hip/hip_runtime.h>
#include <hip/hip_bf16.h>

typedef _Float16 f16;
typedef __attribute__((ext_vector_type(16))) _Float16 v16h;
typedef __attribute__((ext_vector_type(8)))  _Float16 v8h;
typedef __attribute__((ext_vector_type(8)))  float    v8f;

#define T_LEN 2048
#define E_DIM 256
#define H2    256
#define G4    1024      // 4*H2
#define L_LAB 32
#define CLS_T 0
#define SEP_T 31
#define NEGV  (-10000.0f)

// ---------------------------------------------------------------------------
// Prep kernels
// ---------------------------------------------------------------------------
__global__ void embed_gather_f16(const int* __restrict__ ids,
                                 const float* __restrict__ emb,
                                 f16* __restrict__ x16) {
    int t = blockIdx.x, e = threadIdx.x;       // 2048 x 256
    x16[(size_t)t * E_DIM + e] = (f16)emb[(size_t)ids[t] * E_DIM + e];
}

__global__ void cvt_f16(const float* __restrict__ src, f16* __restrict__ dst, int n) {
    int i = blockIdx.x * blockDim.x + threadIdx.x;
    if (i < n) dst[i] = (f16)src[i];
}

__global__ void transpose_whh(const float* __restrict__ w,   // [1024][256]
                              float* __restrict__ wt) {      // [256][1024]
    int j = blockIdx.x, k = threadIdx.x;       // 1024 x 256
    wt[(size_t)k * G4 + j] = w[(size_t)j * E_DIM + k];
}

// ---------------------------------------------------------------------------
// Gate pre-projection: gates[dir][t][j] = x[t] . W_ih[dir][j] + b[dir][j]
// M=2048, N=1024, K=256  -> one 16x16 tile per wave, 8 WMMAs of K=32
// ---------------------------------------------------------------------------
__global__ void gates_gemm(const f16* __restrict__ x16,     // [2048][256]
                           const f16* __restrict__ wih16,   // [2][1024][256]
                           const float* __restrict__ b_f,
                           const float* __restrict__ b_b,
                           float* __restrict__ gates) {     // [2][2048][1024]
    const int lane = threadIdx.x & 31;
    const int w = blockIdx.x * (blockDim.x >> 5) + (threadIdx.x >> 5);
    const int NT = G4 / 16;                 // 64
    const int dir = w / (128 * NT);
    const int rem = w % (128 * NT);
    const int t0 = (rem / NT) * 16;
    const int j0 = (rem % NT) * 16;

    const int mn = lane & 15;               // M for A, N for B (same index)
    const int hi = lane >> 4;

    const f16* __restrict__ arow = x16 + (size_t)(t0 + mn) * E_DIM + hi * 8;
    const f16* __restrict__ brow = wih16 + (size_t)dir * G4 * E_DIM
                                         + (size_t)(j0 + mn) * E_DIM + hi * 8;
    v8f c = {};
#pragma unroll
    for (int kc = 0; kc < 8; ++kc) {
        const int e0 = kc * 32;
        v8h alo = *(const v8h*)(arow + e0);
        v8h ahi = *(const v8h*)(arow + e0 + 16);
        v16h a = __builtin_shufflevector(alo, ahi, 0,1,2,3,4,5,6,7,8,9,10,11,12,13,14,15);
        v8h blo = *(const v8h*)(brow + e0);
        v8h bhi = *(const v8h*)(brow + e0 + 16);
        v16h b = __builtin_shufflevector(blo, bhi, 0,1,2,3,4,5,6,7,8,9,10,11,12,13,14,15);
        c = __builtin_amdgcn_wmma_f32_16x16x32_f16(false, a, false, b,
                                                   (short)0, c, false, false);
    }
    const float bj = (dir ? b_b : b_f)[j0 + mn];
    float* __restrict__ outp = gates + ((size_t)dir * T_LEN + t0 + hi * 8) * G4 + j0 + mn;
#pragma unroll
    for (int v = 0; v < 8; ++v) outp[(size_t)v * G4] = c[v] + bj;
}

// ---------------------------------------------------------------------------
// Sequential bidirectional LSTM recurrence. One block per direction,
// 512 threads; thread owns 2 gates (float2 FMA), W_hh pre-transposed [k][j]
// so loads are fully coalesced and L2-resident across all 2048 steps.
// ---------------------------------------------------------------------------
__global__ void lstm_recur(const float* __restrict__ gates,  // [2][2048][1024]
                           const float* __restrict__ whht,   // [2][256][1024]
                           const float* __restrict__ h0,
                           const float* __restrict__ c0,
                           f16* __restrict__ hs16) {         // [2048][512]
    __shared__ float h_lds[H2];
    __shared__ float zbuf[G4];
    const int dir = blockIdx.x;
    const int tid = threadIdx.x;              // 0..511
    const float* __restrict__ G  = gates + (size_t)dir * T_LEN * G4;
    const float* __restrict__ Wt = whht  + (size_t)dir * H2 * G4 + 2 * tid;

    float c_reg = 0.0f;
    if (tid < H2) {
        h_lds[tid] = h0[dir * H2 + tid];
        c_reg      = c0[dir * H2 + tid];
    }
    __syncthreads();

    for (int step = 0; step < T_LEN; ++step) {
        const int t = dir ? (T_LEN - 1 - step) : step;
        float2 acc = *(const float2*)(G + (size_t)t * G4 + 2 * tid);
#pragma unroll 8
        for (int k = 0; k < H2; ++k) {
            const float hk = h_lds[k];
            const float2 wv = *(const float2*)(Wt + (size_t)k * G4);
            acc.x = fmaf(wv.x, hk, acc.x);
            acc.y = fmaf(wv.y, hk, acc.y);
        }
        zbuf[2 * tid]     = acc.x;
        zbuf[2 * tid + 1] = acc.y;
        __syncthreads();
        if (tid < H2) {
            const float ig = 1.0f / (1.0f + __expf(-zbuf[tid]));
            const float fg = 1.0f / (1.0f + __expf(-zbuf[H2 + tid]));
            const float gg = tanhf(zbuf[2 * H2 + tid]);
            const float og = 1.0f / (1.0f + __expf(-zbuf[3 * H2 + tid]));
            c_reg = fg * c_reg + ig * gg;
            const float h = og * tanhf(c_reg);
            h_lds[tid] = h;
            hs16[(size_t)t * (2 * H2) + dir * H2 + tid] = (f16)h;
        }
        __syncthreads();
    }
}

// ---------------------------------------------------------------------------
// logits[t][l] = hs[t] . w_lin[l] + b_lin[l];  M=2048, N=32, K=512 via WMMA
// ---------------------------------------------------------------------------
__global__ void logits_gemm(const f16* __restrict__ hs16,    // [2048][512]
                            const f16* __restrict__ wlin16,  // [32][512]
                            const float* __restrict__ b_lin,
                            float* __restrict__ logits) {    // [2048][32]
    const int lane = threadIdx.x & 31;
    const int w = blockIdx.x * (blockDim.x >> 5) + (threadIdx.x >> 5);
    const int t0 = (w >> 1) * 16;
    const int j0 = (w & 1) * 16;
    const int mn = lane & 15;
    const int hi = lane >> 4;

    const f16* __restrict__ arow = hs16   + (size_t)(t0 + mn) * (2 * H2) + hi * 8;
    const f16* __restrict__ brow = wlin16 + (size_t)(j0 + mn) * (2 * H2) + hi * 8;
    v8f c = {};
#pragma unroll
    for (int kc = 0; kc < 16; ++kc) {
        const int e0 = kc * 32;
        v8h alo = *(const v8h*)(arow + e0);
        v8h ahi = *(const v8h*)(arow + e0 + 16);
        v16h a = __builtin_shufflevector(alo, ahi, 0,1,2,3,4,5,6,7,8,9,10,11,12,13,14,15);
        v8h blo = *(const v8h*)(brow + e0);
        v8h bhi = *(const v8h*)(brow + e0 + 16);
        v16h b = __builtin_shufflevector(blo, bhi, 0,1,2,3,4,5,6,7,8,9,10,11,12,13,14,15);
        c = __builtin_amdgcn_wmma_f32_16x16x32_f16(false, a, false, b,
                                                   (short)0, c, false, false);
    }
    const float bj = b_lin[j0 + mn];
    float* __restrict__ outp = logits + (size_t)(t0 + hi * 8) * L_LAB + j0 + mn;
#pragma unroll
    for (int v = 0; v < 8; ++v) outp[(size_t)v * L_LAB] = c[v] + bj;
}

// ---------------------------------------------------------------------------
// CRF negative log-likelihood: single wave32, lane n owns state n,
// trans[n][:] register-resident, alpha broadcast by __shfl.
// ---------------------------------------------------------------------------
__global__ void crf_nll(const float* __restrict__ logits,   // [2048][32]
                        const float* __restrict__ trans,    // [32][32]
                        const int* __restrict__ target,     // [2048]
                        float* __restrict__ out) {
    const int lane = threadIdx.x;           // 0..31
    float tr[L_LAB];
#pragma unroll
    for (int p = 0; p < L_LAB; ++p) tr[p] = trans[lane * L_LAB + p];

    float alpha = (lane == CLS_T) ? 0.0f : NEGV;
    for (int t = 0; t < T_LEN; ++t) {
        const float emit = logits[(size_t)t * L_LAB + lane];
        float m = -3.4e38f;
#pragma unroll
        for (int p = 0; p < L_LAB; ++p) {
            const float ap = __shfl(alpha, p, 32);
            m = fmaxf(m, ap + tr[p]);
        }
        float s = 0.0f;
#pragma unroll
        for (int p = 0; p < L_LAB; ++p) {
            const float ap = __shfl(alpha, p, 32);
            s += __expf(ap + tr[p] - m);
        }
        alpha = m + __logf(s) + emit;
    }
    // forward_score = logsumexp(alpha + trans[SEP])
    float v = alpha + trans[SEP_T * L_LAB + lane];
    float m = v;
#pragma unroll
    for (int off = 16; off > 0; off >>= 1) m = fmaxf(m, __shfl_xor(m, off, 32));
    float s = __expf(v - m);
#pragma unroll
    for (int off = 16; off > 0; off >>= 1) s += __shfl_xor(s, off, 32);
    const float fwd = m + __logf(s);

    // gold path score (trans + emit), parallel over lanes
    float acc = 0.0f;
    for (int t = lane; t < T_LEN; t += 32) {
        const int cur  = target[t];
        const int prev = (t == 0) ? CLS_T : target[t - 1];
        acc += trans[cur * L_LAB + prev] + logits[(size_t)t * L_LAB + cur];
    }
#pragma unroll
    for (int off = 16; off > 0; off >>= 1) acc += __shfl_xor(acc, off, 32);

    if (lane == 0) {
        const int last = target[T_LEN - 1];
        out[0] = fwd - (acc + trans[SEP_T * L_LAB + last]);
    }
}

// ---------------------------------------------------------------------------
extern "C" void kernel_launch(void* const* d_in, const int* in_sizes, int n_in,
                              void* d_out, int out_size, void* d_ws, size_t ws_size,
                              hipStream_t stream) {
    const int*   ids    = (const int*)d_in[0];
    const int*   target = (const int*)d_in[2];
    const float* emb    = (const float*)d_in[3];
    const float* w_ih_f = (const float*)d_in[4];
    const float* w_hh_f = (const float*)d_in[5];
    const float* b_f    = (const float*)d_in[6];
    const float* w_ih_b = (const float*)d_in[7];
    const float* w_hh_b = (const float*)d_in[8];
    const float* b_b    = (const float*)d_in[9];
    const float* w_lin  = (const float*)d_in[10];
    const float* b_lin  = (const float*)d_in[11];
    const float* trans  = (const float*)d_in[12];
    const float* h0     = (const float*)d_in[13];
    const float* c0     = (const float*)d_in[14];

    // workspace carve (all 256B aligned)
    char* p = (char*)d_ws;
    auto take = [&](size_t bytes) { char* r = p; p += (bytes + 255) & ~(size_t)255; return r; };
    f16*   x16    = (f16*)  take((size_t)T_LEN * E_DIM * 2);            // 1 MB
    f16*   wih16  = (f16*)  take((size_t)2 * G4 * E_DIM * 2);           // 1 MB
    f16*   wlin16 = (f16*)  take((size_t)L_LAB * 2 * H2 * 2);           // 32 KB
    float* whht   = (float*)take((size_t)2 * H2 * G4 * 4);              // 2 MB
    float* gates  = (float*)take((size_t)2 * T_LEN * G4 * 4);           // 16 MB
    f16*   hs16   = (f16*)  take((size_t)T_LEN * 2 * H2 * 2);           // 2 MB
    float* logits = (float*)take((size_t)T_LEN * L_LAB * 4);            // 256 KB

    // prep: gather + f16 conversions + W_hh transpose
    embed_gather_f16<<<T_LEN, E_DIM, 0, stream>>>(ids, emb, x16);
    cvt_f16<<<(G4 * E_DIM + 255) / 256, 256, 0, stream>>>(w_ih_f, wih16, G4 * E_DIM);
    cvt_f16<<<(G4 * E_DIM + 255) / 256, 256, 0, stream>>>(w_ih_b, wih16 + (size_t)G4 * E_DIM, G4 * E_DIM);
    cvt_f16<<<(L_LAB * 2 * H2 + 255) / 256, 256, 0, stream>>>(w_lin, wlin16, L_LAB * 2 * H2);
    transpose_whh<<<G4, H2, 0, stream>>>(w_hh_f, whht);
    transpose_whh<<<G4, H2, 0, stream>>>(w_hh_b, whht + (size_t)H2 * G4);

    // time-parallel input projection (WMMA): 2 x [2048,1024] = x @ W_ih^T + b
    gates_gemm<<<(2 * 128 * 64) / 8, 256, 0, stream>>>(x16, wih16, b_f, b_b, gates);

    // sequential recurrence (the true critical path)
    lstm_recur<<<2, 512, 0, stream>>>(gates, whht, h0, c0, hs16);

    // classifier (WMMA): [2048,32] = hs @ w_lin^T + b_lin
    logits_gemm<<<(128 * 2) / 8, 256, 0, stream>>>(hs16, wlin16, b_lin, logits);

    // CRF scan -> scalar
    crf_nll<<<1, 32, 0, stream>>>(logits, trans, target, (float*)d_out);
}
